// RotationNet_41437844472400
// MI455X (gfx1250) — compile-verified, matrix operand order checked by reading, not run on hardware
//
#include <hip/hip_runtime.h>
#include <stdint.h>
#include <stddef.h>

// ---------------------------------------------------------------------------
// RotationNet for MI455X (gfx1250, wave32).
// Fused dw+pw+ELU blocks; bf16 activations; pointwise convs via
// v_wmma_f32_16x16x32_bf16 (16 px x 64 ch tile per wave).
// ---------------------------------------------------------------------------

typedef __bf16 v16bf __attribute__((ext_vector_type(16)));
typedef float  v8f   __attribute__((ext_vector_type(8)));

#define B_  32
#define H0 480
#define W0 640
#define H1 240
#define W1 320
#define C1 32
#define H2 120
#define W2 160
#define C2 64
#define H3 60
#define W3 80
#define C3 64
#define HW3 (H3*W3)   // 4800

__device__ __forceinline__ unsigned short f2bf(float f) {
  unsigned u = __builtin_bit_cast(unsigned, f);
  u += 0x7FFFu + ((u >> 16) & 1u);          // round-to-nearest-even
  return (unsigned short)(u >> 16);
}
__device__ __forceinline__ float bf2f(unsigned short b) {
  return __builtin_bit_cast(float, ((unsigned)b) << 16);
}
__device__ __forceinline__ float elu1(float x) {
  return x > 0.f ? x : (__expf(x) - 1.f);
}
// Build a WMMA 16-bit fragment (16 bf16 lanes-worth) from two 16B chunks.
__device__ __forceinline__ v16bf mk_frag(uint4 a, uint4 b) {
  unsigned v[8] = {a.x, a.y, a.z, a.w, b.x, b.y, b.z, b.w};
  v16bf f;
#pragma unroll
  for (int i = 0; i < 8; ++i) {
    f[2*i]   = __builtin_bit_cast(__bf16, (unsigned short)(v[i] & 0xFFFFu));
    f[2*i+1] = __builtin_bit_cast(__bf16, (unsigned short)(v[i] >> 16));
  }
  return f;
}

// ---------------------------------------------------------------------------
// k0: pack pointwise weights fp32 -> bf16 (consumed as WMMA A matrices).
// ---------------------------------------------------------------------------
__global__ void k0_pack(const float* __restrict__ pw2w, const float* __restrict__ pw3w,
                        unsigned short* __restrict__ pw2bf, unsigned short* __restrict__ pw3bf) {
  int t = blockIdx.x * blockDim.x + threadIdx.x;
  if (t < C2 * C1) pw2bf[t] = f2bf(pw2w[t]);
  int u = t - C2 * C1;
  if (u >= 0 && u < C3 * C2) pw3bf[u] = f2bf(pw3w[u]);
}

// ---------------------------------------------------------------------------
// k1: fused coord-gen + depthwise(5ch,3x3,s2) + pointwise(5->32) + ELU.
// One thread per output pixel, writes 32 bf16 channels (64B) NHWC.
// ---------------------------------------------------------------------------
__global__ void __launch_bounds__(256)
k1_block1(const float* __restrict__ flow, const float* __restrict__ dw1,
          const float* __restrict__ pw1w, const float* __restrict__ pw1b,
          unsigned short* __restrict__ f1) {
  int t = blockIdx.x * blockDim.x + threadIdx.x;
  if (t >= B_ * H1 * W1) return;
  int x = t % W1;
  int y = (t / W1) % H1;
  int b = t / (W1 * H1);

  float d[5] = {0.f, 0.f, 0.f, 0.f, 0.f};
#pragma unroll
  for (int ky = 0; ky < 3; ++ky) {
    int iy = 2 * y - 1 + ky;
    if (iy < 0 || iy >= H0) continue;
    float yv = -1.f + 2.f * (float)iy / (float)(H0 - 1);
#pragma unroll
    for (int kx = 0; kx < 3; ++kx) {
      int ix = 2 * x - 1 + kx;
      if (ix < 0 || ix >= W0) continue;
      float xv = -1.f + 2.f * (float)ix / (float)(W0 - 1);
      int k = ky * 3 + kx;
      float fA = flow[((size_t)(b * 2 + 0) * H0 + iy) * W0 + ix];
      float fB = flow[((size_t)(b * 2 + 1) * H0 + iy) * W0 + ix];
      d[0] += dw1[0 * 9 + k] * fA;
      d[1] += dw1[1 * 9 + k] * fB;
      d[2] += dw1[2 * 9 + k] * xv;
      d[3] += dw1[3 * 9 + k] * yv;
      d[4] += dw1[4 * 9 + k] * (xv * xv + yv * yv);
    }
  }
  alignas(16) unsigned short o[C1];
#pragma unroll
  for (int oc = 0; oc < C1; ++oc) {
    float a = pw1b[oc];
#pragma unroll
    for (int ic = 0; ic < 5; ++ic) a += pw1w[oc * 5 + ic] * d[ic];
    o[oc] = f2bf(elu1(a));
  }
  uint4* dst = (uint4*)&f1[(size_t)t * C1];
  const uint4* src = (const uint4*)o;
  dst[0] = src[0]; dst[1] = src[1]; dst[2] = src[2]; dst[3] = src[3];
}

// ---------------------------------------------------------------------------
// k2: fused depthwise(32ch,3x3,s2) + WMMA pointwise(32->64) + ELU.
// One wave per 16-pixel tile; 4x v_wmma_f32_16x16x32_bf16.
// ---------------------------------------------------------------------------
__global__ void __launch_bounds__(32)
k2_block2(const unsigned short* __restrict__ f1, const float* __restrict__ dw2,
          const unsigned short* __restrict__ pw2bf, const float* __restrict__ pw2b,
          unsigned short* __restrict__ f2) {
  __shared__ unsigned short tile[16 * C1];   // 16 px x 32 ch bf16 (1 KB)
  int l = threadIdx.x;
  int x0 = blockIdx.x * 16, y = blockIdx.y, b = blockIdx.z;

  __builtin_prefetch(&pw2bf[(l & 15) * C1], 0, 0);

  // depthwise: each lane computes 16 channels of one pixel
  int p = l >> 1, c0 = (l & 1) * 16;
  float acc[16];
#pragma unroll
  for (int i = 0; i < 16; ++i) acc[i] = 0.f;
#pragma unroll
  for (int ky = 0; ky < 3; ++ky) {
    int iy = 2 * y - 1 + ky;
    if (iy < 0 || iy >= H1) continue;
#pragma unroll
    for (int kx = 0; kx < 3; ++kx) {
      int ix = 2 * (x0 + p) - 1 + kx;
      if (ix < 0 || ix >= W1) continue;
      const uint4* q = (const uint4*)&f1[(((size_t)b * H1 + iy) * W1 + ix) * C1 + c0];
      uint4 q0 = q[0], q1 = q[1];
      unsigned v[8] = {q0.x, q0.y, q0.z, q0.w, q1.x, q1.y, q1.z, q1.w};
      int k = ky * 3 + kx;
#pragma unroll
      for (int i = 0; i < 8; ++i) {
        acc[2*i]   += dw2[(c0 + 2*i)     * 9 + k] * bf2f((unsigned short)(v[i] & 0xFFFFu));
        acc[2*i+1] += dw2[(c0 + 2*i + 1) * 9 + k] * bf2f((unsigned short)(v[i] >> 16));
      }
    }
  }
#pragma unroll
  for (int i = 0; i < 16; ++i) tile[p * C1 + c0 + i] = f2bf(acc[i]);
  __syncthreads();

  // WMMA: A = pw2 weights (64x32 bf16), B = tile (K=32 ch x N=16 px)
  int pb = l & 15;
  int koff = (l < 16) ? 0 : 8;         // per-ISA K split across lane halves
  uint4 b0 = *(const uint4*)&tile[pb * C1 + koff];
  uint4 b1 = *(const uint4*)&tile[pb * C1 + koff + 16];
  v16bf bfrag = mk_frag(b0, b1);

  int row = l & 15;
  v16bf afrag[4];
  v8f accv[4];
#pragma unroll
  for (int mt = 0; mt < 4; ++mt) {
    const unsigned short* ar = &pw2bf[(mt * 16 + row) * C1];
    afrag[mt] = mk_frag(*(const uint4*)&ar[koff], *(const uint4*)&ar[koff + 16]);
#pragma unroll
    for (int r = 0; r < 8; ++r) accv[mt][r] = pw2b[mt * 16 + koff + r];  // bias in C
  }
#pragma unroll
  for (int mt = 0; mt < 4; ++mt)
    accv[mt] = __builtin_amdgcn_wmma_f32_16x16x32_bf16(
        false, afrag[mt], false, bfrag, (short)0, accv[mt], false, false);

  size_t obase = (((size_t)b * H2 + y) * W2 + x0 + pb) * C2;
#pragma unroll
  for (int mt = 0; mt < 4; ++mt) {
    alignas(16) unsigned short o[8];
#pragma unroll
    for (int r = 0; r < 8; ++r) o[r] = f2bf(elu1(accv[mt][r]));
    *(uint4*)&f2[obase + mt * 16 + koff] = *(const uint4*)o;  // 8 consecutive channels
  }
}

// ---------------------------------------------------------------------------
// k3: fused depthwise(64ch,3x3,s2) + WMMA pointwise(64->64) + ELU
//     + attention logits (f . attn_w + attn_b). 8x WMMA per tile (K=64).
// ---------------------------------------------------------------------------
__global__ void __launch_bounds__(32)
k3_block3(const unsigned short* __restrict__ f2, const float* __restrict__ dw3,
          const unsigned short* __restrict__ pw3bf, const float* __restrict__ pw3b,
          const float* __restrict__ attnw, const float* __restrict__ attnb,
          float* __restrict__ f3, float* __restrict__ logits) {
  __shared__ unsigned short tile[16 * C2];   // 16 px x 64 ch bf16 (2 KB)
  int l = threadIdx.x;
  int x0 = blockIdx.x * 16, y = blockIdx.y, b = blockIdx.z;

  int p = l >> 1, c0 = (l & 1) * 32;
  float acc[32];
#pragma unroll
  for (int i = 0; i < 32; ++i) acc[i] = 0.f;
#pragma unroll
  for (int ky = 0; ky < 3; ++ky) {
    int iy = 2 * y - 1 + ky;
    if (iy < 0 || iy >= H2) continue;
#pragma unroll
    for (int kx = 0; kx < 3; ++kx) {
      int ix = 2 * (x0 + p) - 1 + kx;
      if (ix < 0 || ix >= W2) continue;
      const uint4* q = (const uint4*)&f2[(((size_t)b * H2 + iy) * W2 + ix) * C2 + c0];
      int k = ky * 3 + kx;
#pragma unroll
      for (int j = 0; j < 4; ++j) {
        uint4 qq = q[j];
        unsigned v[4] = {qq.x, qq.y, qq.z, qq.w};
#pragma unroll
        for (int i = 0; i < 4; ++i) {
          int c = j * 8 + 2 * i;
          acc[c]     += dw3[(c0 + c)     * 9 + k] * bf2f((unsigned short)(v[i] & 0xFFFFu));
          acc[c + 1] += dw3[(c0 + c + 1) * 9 + k] * bf2f((unsigned short)(v[i] >> 16));
        }
      }
    }
  }
#pragma unroll
  for (int i = 0; i < 32; ++i) tile[p * C2 + c0 + i] = f2bf(acc[i]);
  __syncthreads();

  int pb = l & 15;
  int koff = (l < 16) ? 0 : 8;
  int row = l & 15;

  v16bf bfrag[2];
#pragma unroll
  for (int kt = 0; kt < 2; ++kt)
    bfrag[kt] = mk_frag(*(const uint4*)&tile[pb * C2 + kt * 32 + koff],
                        *(const uint4*)&tile[pb * C2 + kt * 32 + koff + 16]);

  v8f accv[4];
#pragma unroll
  for (int mt = 0; mt < 4; ++mt)
#pragma unroll
    for (int r = 0; r < 8; ++r) accv[mt][r] = pw3b[mt * 16 + koff + r];

#pragma unroll
  for (int mt = 0; mt < 4; ++mt) {
#pragma unroll
    for (int kt = 0; kt < 2; ++kt) {
      const unsigned short* ar = &pw3bf[(mt * 16 + row) * C2 + kt * 32];
      v16bf afrag = mk_frag(*(const uint4*)&ar[koff], *(const uint4*)&ar[koff + 16]);
      accv[mt] = __builtin_amdgcn_wmma_f32_16x16x32_bf16(
          false, afrag, false, bfrag[kt], (short)0, accv[mt], false, false);
    }
  }

  // ELU, store f32 NHWC, and fused attention logit partial
  float lpart = 0.f;
  size_t obase = (((size_t)b * H3 + y) * W3 + x0 + pb) * C3;
#pragma unroll
  for (int mt = 0; mt < 4; ++mt) {
    alignas(32) float ov[8];
#pragma unroll
    for (int r = 0; r < 8; ++r) {
      float v = elu1(accv[mt][r]);
      ov[r] = v;
      lpart += attnw[mt * 16 + koff + r] * v;
    }
    float4* dst = (float4*)&f3[obase + mt * 16 + koff];
    dst[0] = *(const float4*)&ov[0];
    dst[1] = *(const float4*)&ov[4];
  }
  lpart += __shfl_xor(lpart, 16);      // lanes l and l+16 hold same pixel
  if (l < 16) logits[((size_t)b * H3 + y) * W3 + x0 + pb] = lpart + attnb[0];
}

// ---------------------------------------------------------------------------
// k4: per-batch softmax attention pooling (logits in LDS) + MLP + GRU + out.
// f3 is L2-resident (39 MB << 192 MB L2).
// ---------------------------------------------------------------------------
__global__ void __launch_bounds__(256)
k4_head(const float* __restrict__ f3, const float* __restrict__ logits,
        const float* __restrict__ mlp1w, const float* __restrict__ mlp1b,
        const float* __restrict__ mlp2w, const float* __restrict__ mlp2b,
        const float* __restrict__ w_ih, const float* __restrict__ b_ih,
        const float* __restrict__ b_hh,
        const float* __restrict__ fc_w, const float* __restrict__ fc_b,
        float* __restrict__ out) {
  __shared__ float lg[HW3];       // 19.2 KB softmax weights
  __shared__ float red[256];
  __shared__ float pp[4 * 64];
  __shared__ float sm[64 + 32 + 3 + 32];
  int t = threadIdx.x, b = blockIdx.x;
  const float* fb = f3 + (size_t)b * HW3 * C3;

  float mx = -1e30f;
  for (int p = t; p < HW3; p += 256) {
    float v = logits[(size_t)b * HW3 + p];
    lg[p] = v;
    mx = fmaxf(mx, v);
  }
  red[t] = mx; __syncthreads();
  for (int s = 128; s > 0; s >>= 1) { if (t < s) red[t] = fmaxf(red[t], red[t + s]); __syncthreads(); }
  mx = red[0]; __syncthreads();

  float sum = 0.f;
  for (int p = t; p < HW3; p += 256) { float e = __expf(lg[p] - mx); lg[p] = e; sum += e; }
  red[t] = sum; __syncthreads();
  for (int s = 128; s > 0; s >>= 1) { if (t < s) red[t] += red[t + s]; __syncthreads(); }
  float inv = 1.f / red[0];

  // pooled[c] = sum_p w_p * f[p][c]
  int c = t & 63, g = t >> 6;
  float part = 0.f;
  for (int p = g; p < HW3; p += 4) part += lg[p] * fb[(size_t)p * C3 + c];
  pp[g * 64 + c] = part;
  __syncthreads();

  float* pooled = sm;         // [64]
  float* h1 = sm + 64;        // [32]
  float* om = sm + 96;        // [3]
  float* hh = sm + 99;        // [32]
  if (t < 64) pooled[t] = (pp[t] + pp[64 + t] + pp[128 + t] + pp[192 + t]) * inv;
  __syncthreads();

  if (t < 32) {
    float a = mlp1b[t];
    for (int k = 0; k < 64; ++k) a += mlp1w[t * 64 + k] * pooled[k];
    h1[t] = elu1(a);
  }
  __syncthreads();
  if (t < 3) {
    float a = mlp2b[t];
    for (int k = 0; k < 32; ++k) a += mlp2w[t * 32 + k] * h1[k];
    om[t] = a;
  }
  __syncthreads();
  if (t < 32) {                 // GRU step, h0 = 0 -> gh = b_hh
    int j = t;
    float ir = b_ih[j], iz = b_ih[32 + j], in_ = b_ih[64 + j];
    for (int k = 0; k < 3; ++k) {
      ir  += w_ih[j * 3 + k]        * om[k];
      iz  += w_ih[(32 + j) * 3 + k] * om[k];
      in_ += w_ih[(64 + j) * 3 + k] * om[k];
    }
    float r = 1.f / (1.f + __expf(-(ir + b_hh[j])));
    float z = 1.f / (1.f + __expf(-(iz + b_hh[32 + j])));
    float n = tanhf(in_ + r * b_hh[64 + j]);
    hh[j] = (1.f - z) * n;
  }
  __syncthreads();
  if (t < 3) {
    float a = fc_b[t] + om[t];
    for (int k = 0; k < 32; ++k) a += fc_w[t * 32 + k] * hh[k];
    out[b * 3 + t] = a;
  }
}

// ---------------------------------------------------------------------------
extern "C" void kernel_launch(void* const* d_in, const int* in_sizes, int n_in,
                              void* d_out, int out_size, void* d_ws, size_t ws_size,
                              hipStream_t stream) {
  const float* flow  = (const float*)d_in[0];
  const float* dw1   = (const float*)d_in[1];
  const float* pw1w  = (const float*)d_in[2];
  const float* pw1b  = (const float*)d_in[3];
  const float* dw2   = (const float*)d_in[4];
  const float* pw2w  = (const float*)d_in[5];
  const float* pw2b  = (const float*)d_in[6];
  const float* dw3   = (const float*)d_in[7];
  const float* pw3w  = (const float*)d_in[8];
  const float* pw3b  = (const float*)d_in[9];
  const float* attnw = (const float*)d_in[10];
  const float* attnb = (const float*)d_in[11];
  const float* mlp1w = (const float*)d_in[12];
  const float* mlp1b = (const float*)d_in[13];
  const float* mlp2w = (const float*)d_in[14];
  const float* mlp2b = (const float*)d_in[15];
  const float* w_ih  = (const float*)d_in[16];
  const float* b_ih  = (const float*)d_in[18];
  const float* b_hh  = (const float*)d_in[19];
  const float* fc_w  = (const float*)d_in[20];
  const float* fc_b  = (const float*)d_in[21];
  float* out = (float*)d_out;

  char* ws = (char*)d_ws;
  const size_t OFF_PW2 = 0;                                   // 4 KB
  const size_t OFF_PW3 = 4096;                                // 8 KB
  const size_t OFF_F1  = 12288;                               // 157.3 MB bf16
  const size_t OFF_F2  = OFF_F1 + (size_t)B_*H1*W1*C1*2;      // 78.6 MB bf16
  const size_t OFF_F3  = OFF_F2 + (size_t)B_*H2*W2*C2*2;      // 39.3 MB f32
  const size_t OFF_LG  = OFF_F3 + (size_t)B_*H3*W3*C3*4;      // 0.6 MB f32
  unsigned short* pw2bf = (unsigned short*)(ws + OFF_PW2);
  unsigned short* pw3bf = (unsigned short*)(ws + OFF_PW3);
  unsigned short* f1    = (unsigned short*)(ws + OFF_F1);
  unsigned short* f2    = (unsigned short*)(ws + OFF_F2);
  float* f3             = (float*)(ws + OFF_F3);
  float* lgbuf          = (float*)(ws + OFF_LG);

  k0_pack<<<(C2*C1 + C3*C2 + 255) / 256, 256, 0, stream>>>(pw2w, pw3w, pw2bf, pw3bf);

  int n1 = B_ * H1 * W1;
  k1_block1<<<(n1 + 255) / 256, 256, 0, stream>>>(flow, dw1, pw1w, pw1b, f1);

  k2_block2<<<dim3(W2 / 16, H2, B_), 32, 0, stream>>>(f1, dw2, pw2bf, pw2b, f2);

  k3_block3<<<dim3(W3 / 16, H3, B_), 32, 0, stream>>>(f2, dw3, pw3bf, pw3b,
                                                      attnw, attnb, f3, lgbuf);

  k4_head<<<B_, 256, 0, stream>>>(f3, lgbuf, mlp1w, mlp1b, mlp2w, mlp2b,
                                  w_ih, b_ih, b_hh, fc_w, fc_b, out);
}
